// SwinTransformerBlock_85718957293905
// MI455X (gfx1250) — matrix operand order repaired and checked by reading, not compiled, not run on hardware
//
#include <hip/hip_runtime.h>
#include <hip/hip_bf16.h>

// ---------------------------------------------------------------- constants
#define Bn     2
#define Cc     192
#define HEADS  6
#define HD     32
#define WSz    7
#define SHIFTc 3
#define Sdim   28
#define Ltok   (28*28*28)        // 21952
#define Nwin   343               // tokens per window
#define NWl    64                // windows per image
#define Mtok   (Bn*Ltok)         // 43904 == 128*343, divisible by 16
#define HF     768
#define BNc    24
#define KCONV  672               // 648 padded to multiple of 32
#define NPAD   352               // 343 padded to 22*16
#define SCALEq 0.17677669529663689f   // 32^-0.5

typedef __attribute__((ext_vector_type(16))) _Float16 v16h;
typedef __attribute__((ext_vector_type(8)))  _Float16 v8h;
typedef __attribute__((ext_vector_type(8)))  float    v8f;

// ------------------------------------------------ WMMA fragment load helpers
// A (16x32 f16): lane l -> row (l&15); halves: K = koff..koff+7, koff+16..koff+23,
// koff = (l>>4)*8   (ISA 7.12.2, 16-bit A layout)
__device__ __forceinline__ v16h load_a_frag(const _Float16* base, int ld) {
  const int l = threadIdx.x & 31;
  const _Float16* q = base + (size_t)(l & 15) * ld + ((l >> 4) << 3);
  v16h f;
  ((v8h*)&f)[0] = *(const v8h*)(q);
  ((v8h*)&f)[1] = *(const v8h*)(q + 16);
  return f;
}
// B (32x16 f16) with B[k][n] = W[n][k], W row-major [N][K]:
// lane l -> column n = (l&15); K = koff..koff+7, koff+16..koff+23, koff = (l>>4)*8
__device__ __forceinline__ v16h load_b_frag(const _Float16* base, int ld) {
  const int l = threadIdx.x & 31;
  const _Float16* q = base + (size_t)(l & 15) * ld + ((l >> 4) << 3);
  v16h f;
  ((v8h*)&f)[0] = *(const v8h*)(q);
  ((v8h*)&f)[1] = *(const v8h*)(q + 16);
  return f;
}
__device__ __forceinline__ v8f wmma32(v16h a, v16h b, v8f c) {
  return __builtin_amdgcn_wmma_f32_16x16x32_f16(false, a, false, b, (short)0, c,
                                                false, false);
}

// ---------------------------------------------------------------- weight packs
__global__ void pack_dense(const float* __restrict__ src, _Float16* __restrict__ dst,
                           int N, int Nround, int K) {
  int tid = blockIdx.x * blockDim.x + threadIdx.x;
  int total = Nround * K;
  if (tid >= total) return;
  int k = tid % K, o = tid / K;
  dst[tid] = (o < N) ? (_Float16)src[(size_t)o * K + k] : (_Float16)0.f;
}
// conv weight (O,24,3,3,3) -> [Oround][672] with col = j*24+c, zero padded
__global__ void pack_conv(const float* __restrict__ src, _Float16* __restrict__ dst,
                          int O, int Oround) {
  int tid = blockIdx.x * blockDim.x + threadIdx.x;
  int total = Oround * KCONV;
  if (tid >= total) return;
  int kk = tid % KCONV, o = tid / KCONV;
  _Float16 v = (_Float16)0.f;
  if (o < O && kk < 648) {
    int j = kk / 24, c = kk % 24;
    v = (_Float16)src[((size_t)o * 24 + c) * 27 + j];
  }
  dst[tid] = v;
}

// ------------------------------------------------------------ generic WMMA GEMM
// out[m][n] = sum_k A[m][k]*W[n][k] + bias[n]  (+resid)   A,W f16 row-major
// EPI: 0 = f32 out, 1 = f16 out, 2 = f32 out + resid add
template <int EPI>
__global__ __launch_bounds__(256) void gemm_xwt(
    const _Float16* __restrict__ A, int lda,
    const _Float16* __restrict__ W, int ldw,
    const float* __restrict__ bias, const float* __restrict__ resid,
    float* __restrict__ outF, _Float16* __restrict__ outH,
    int ldc, int M, int Nstore, int tilesN, int K) {
  int wave = (int)((blockIdx.x * blockDim.x + threadIdx.x) >> 5);
  int nStrips = (tilesN + 3) >> 2;
  int tilesM = M >> 4;
  if (wave >= tilesM * nStrips) return;
  int tm = wave / nStrips, ts = wave % nStrips;
  const int l = threadIdx.x & 31, col = l & 15, rbase = (l >> 4) * 8;
  v8f acc[4] = {};
  for (int kk = 0; kk < K; kk += 32) {
    const _Float16* abase = A + (size_t)(tm * 16) * lda + kk;
    v16h a = load_a_frag(abase, lda);
    if (kk + 32 < K) {
      // next k-step A tile -> global_prefetch_b8 (per-lane row prefetch)
      __builtin_prefetch(abase + 32 + (size_t)(l & 15) * lda, 0, 0);
    }
#pragma unroll
    for (int s = 0; s < 4; ++s) {
      int tn = ts * 4 + s;
      if (tn < tilesN) {
        v16h b = load_b_frag(W + (size_t)(tn * 16) * ldw + kk, ldw);
        acc[s] = wmma32(a, b, acc[s]);
      }
    }
  }
#pragma unroll
  for (int s = 0; s < 4; ++s) {
    int tn = ts * 4 + s;
    if (tn >= tilesN) continue;
    int n = tn * 16 + col;
    if (n >= Nstore) continue;
    float bv = bias ? bias[n] : 0.f;
#pragma unroll
    for (int j = 0; j < 8; ++j) {
      int m = tm * 16 + rbase + j;
      float v = acc[s][j] + bv;
      size_t off = (size_t)m * ldc + n;
      if (EPI == 2) v += resid[off];
      if (EPI == 1) outH[off] = (_Float16)v; else outF[off] = v;
    }
  }
}

// --------------------------------------------------- LN1 + shift + window (f16)
__global__ __launch_bounds__(256) void ln1_window(
    const float* __restrict__ x, const float* __restrict__ g,
    const float* __restrict__ bta, _Float16* __restrict__ xw) {
  int row = (int)((blockIdx.x * blockDim.x + threadIdx.x) >> 5);
  if (row >= Mtok) return;
  const int l = threadIdx.x & 31;
  int w = row / Nwin, n = row % Nwin;
  int b = w >> 6, wl = w & 63;
  int wz = wl >> 4, wy = (wl >> 2) & 3, wx = wl & 3;
  int iz = n / 49, iy = (n / 7) % 7, ix = n % 7;
  int z = (wz * 7 + iz + SHIFTc) % 28;
  int y = (wy * 7 + iy + SHIFTc) % 28;
  int xx = (wx * 7 + ix + SHIFTc) % 28;
  const float* src = x + ((size_t)b * Ltok + (z * 28 + y) * 28 + xx) * Cc;
  float vals[6]; float s = 0.f;
#pragma unroll
  for (int t = 0; t < 6; ++t) { vals[t] = src[l * 6 + t]; s += vals[t]; }
  for (int d = 16; d >= 1; d >>= 1) s += __shfl_xor(s, d, 32);
  float mean = s / 192.f, vs = 0.f;
#pragma unroll
  for (int t = 0; t < 6; ++t) { float dd = vals[t] - mean; vs += dd * dd; }
  for (int d = 16; d >= 1; d >>= 1) vs += __shfl_xor(vs, d, 32);
  float inv = rsqrtf(vs / 192.f + 1e-5f);
  _Float16* dst = xw + (size_t)row * Cc;
#pragma unroll
  for (int t = 0; t < 6; ++t) {
    int c = l * 6 + t;
    dst[c] = (_Float16)((vals[t] - mean) * inv * g[c] + bta[c]);
  }
}

// ------------------------------------------------------------------ plain LN
__global__ __launch_bounds__(256) void ln_rows(
    const float* __restrict__ x, const float* __restrict__ g,
    const float* __restrict__ bta, _Float16* __restrict__ out) {
  int row = (int)((blockIdx.x * blockDim.x + threadIdx.x) >> 5);
  if (row >= Mtok) return;
  const int l = threadIdx.x & 31;
  const float* src = x + (size_t)row * Cc;
  float vals[6]; float s = 0.f;
#pragma unroll
  for (int t = 0; t < 6; ++t) { vals[t] = src[l * 6 + t]; s += vals[t]; }
  for (int d = 16; d >= 1; d >>= 1) s += __shfl_xor(s, d, 32);
  float mean = s / 192.f, vs = 0.f;
#pragma unroll
  for (int t = 0; t < 6; ++t) { float dd = vals[t] - mean; vs += dd * dd; }
  for (int d = 16; d >= 1; d >>= 1) vs += __shfl_xor(vs, d, 32);
  float inv = rsqrtf(vs / 192.f + 1e-5f);
  _Float16* dst = out + (size_t)row * Cc;
#pragma unroll
  for (int t = 0; t < 6; ++t) {
    int c = l * 6 + t;
    dst[c] = (_Float16)((vals[t] - mean) * inv * g[c] + bta[c]);
  }
}

// ---------------------------------------------------- fused windowed attention
// one workgroup per (window, head); 3 waves; 16-row tiles with edge clamp.
// V is staged into LDS row-major with async global->LDS b128 copies, then
// transposed in LDS into the WMMA B-fragment-friendly layout.
__global__ __launch_bounds__(96) void attn_win(
    const _Float16* __restrict__ qkv, const float* __restrict__ rel_tab,
    const float* __restrict__ mask, _Float16* __restrict__ o) {
  __shared__ __align__(16) _Float16 Vt[32 * NPAD];        // [d][slot]
  __shared__ __align__(16) _Float16 Pb[3][16 * NPAD];     // per-wave P strip
  const int w = blockIdx.x / HEADS, h = blockIdx.x % HEADS;
  const int wi = w & 63;
  const size_t rowbase = (size_t)w * Nwin;
  const int qoff = h * 32, koff = 192 + h * 32, voff = 384 + h * 32;

  // ---- stage V rows [slot][32] into scratch LDS (reusing P area) via
  //      global_load_async_to_lds_b128; slot s -> key min((s>>4)*16,327)+(s&15)
  {
    _Float16* Vs = &Pb[0][0];                    // 352*32 halves = 22528 B
    unsigned ldsbase = (unsigned)(uintptr_t)Vs;  // LDS aperture: low 32 bits
    for (int t = threadIdx.x; t < NPAD * 4; t += blockDim.x) {
      int s = t >> 2, c = t & 3;                 // key slot, 16B chunk
      int n0 = (s >> 4) * 16; if (n0 > 327) n0 = 327;
      int kt = n0 + (s & 15);
      const _Float16* gp = qkv + (rowbase + kt) * 576 + voff + c * 8;
      unsigned long long ga = (unsigned long long)(uintptr_t)gp;
      unsigned la = ldsbase + (unsigned)(s * 64 + c * 16);
      asm volatile("global_load_async_to_lds_b128 %0, %1, off"
                   :: "v"(la), "v"(ga) : "memory");
    }
    asm volatile("s_wait_asynccnt 0" ::: "memory");
    __syncthreads();
    // transpose Vs[slot][d] -> Vt[d][slot]
    for (int idx = threadIdx.x; idx < 32 * NPAD; idx += blockDim.x) {
      int d = idx / NPAD, s = idx % NPAD;
      Vt[idx] = Vs[s * 32 + d];
    }
  }
  __syncthreads();   // Vt ready; Pb free for per-wave P strips

  const int wave = threadIdx.x >> 5;
  const int l = threadIdx.x & 31, col = l & 15, rsel = l >> 4;
  _Float16* P = Pb[wave];
  for (int rt = wave; rt < 22; rt += 3) {
    int m0 = rt * 16; if (m0 > 327) m0 = 327;
    // scores: S = Q K^T  (Q,K straight from global qkv; scale folded below)
    v16h aq = load_a_frag(qkv + (rowbase + m0) * 576 + qoff, 576);
    v8f sc[22];
#pragma unroll
    for (int ct = 0; ct < 22; ++ct) {
      int n0 = ct * 16; if (n0 > 327) n0 = 327;
      v16h bk = load_b_frag(qkv + (rowbase + n0) * 576 + koff, 576);
      v8f z = {};
      sc[ct] = wmma32(aq, bk, z);
    }
    // epilogue: scale + rel-pos bias + shift mask (+ kill duplicate edge cols)
#pragma unroll
    for (int ct = 0; ct < 22; ++ct) {
      int n0 = ct * 16; if (n0 > 327) n0 = 327;
      int n = n0 + col;
      bool dup = (ct == 21) && (col < 9);
      int zn = n / 49, yn = (n / 7) % 7, xn = n % 7;
#pragma unroll
      for (int j = 0; j < 8; ++j) {
        int m = m0 + rsel * 8 + j;
        float v = sc[ct][j] * SCALEq;
        int zm = m / 49, ym = (m / 7) % 7, xm = m % 7;
        int ridx = (zm - zn + 6) * 169 + (ym - yn + 6) * 13 + (xm - xn + 6);
        v += rel_tab[ridx * 6 + h];
        v += mask[((size_t)wi * Nwin + m) * Nwin + n];
        if (dup) v = -1e30f;
        sc[ct][j] = v;
      }
    }
    // row softmax (rows live on a fixed 16-lane half; reduce with xor shuffles)
#pragma unroll
    for (int j = 0; j < 8; ++j) {
      float mx = -1e30f;
#pragma unroll
      for (int ct = 0; ct < 22; ++ct) mx = fmaxf(mx, sc[ct][j]);
      for (int d = 8; d >= 1; d >>= 1) mx = fmaxf(mx, __shfl_xor(mx, d, 32));
      float sm = 0.f;
#pragma unroll
      for (int ct = 0; ct < 22; ++ct) {
        float e = __expf(sc[ct][j] - mx);
        sc[ct][j] = e; sm += e;
      }
      for (int d = 8; d >= 1; d >>= 1) sm += __shfl_xor(sm, d, 32);
      float r = 1.f / sm;
#pragma unroll
      for (int ct = 0; ct < 22; ++ct)
        P[(rsel * 8 + j) * NPAD + ct * 16 + col] = (_Float16)(sc[ct][j] * r);
    }
    // O = P @ V
#pragma unroll
    for (int c = 0; c < 2; ++c) {
      v8f oacc = {};
#pragma unroll
      for (int i = 0; i < 11; ++i) {
        v16h ap = load_a_frag(P + 32 * i, NPAD);
        v16h bv = load_b_frag(Vt + (size_t)(c * 16) * NPAD + 32 * i, NPAD);
        oacc = wmma32(ap, bv, oacc);
      }
#pragma unroll
      for (int j = 0; j < 8; ++j) {
        int m = m0 + rsel * 8 + j;
        o[(rowbase + m) * Cc + h * 32 + c * 16 + col] = (_Float16)oacc[j];
      }
    }
  }
}

// ---------------------------------------- window reverse + unshift + residual
__global__ void unwindow_resid(const float* __restrict__ x,
                               const float* __restrict__ proj,
                               float* __restrict__ resid) {
  size_t i = blockIdx.x * (size_t)blockDim.x + threadIdx.x;
  size_t total = (size_t)Mtok * Cc;
  if (i >= total) return;
  int c = (int)(i % Cc); size_t t = i / Cc;
  int ll = (int)(t % Ltok); int b = (int)(t / Ltok);
  int z = ll / 784, y = (ll / 28) % 28, xx = ll % 28;
  int pz = (z + 25) % 28, py = (y + 25) % 28, px = (xx + 25) % 28;
  int wz = pz / 7, iz = pz % 7, wy = py / 7, iy = py % 7, wx = px / 7, ix = px % 7;
  int wl = (wz * 4 + wy) * 4 + wx;
  size_t rw = ((size_t)(b * 64 + wl)) * Nwin + (iz * 49 + iy * 7 + ix);
  resid[i] = x[i] + proj[rw * Cc + c];
}

// ------------------------------------------------------- 3^3 avg pool (pad 1)
__global__ void avgpool3(const _Float16* __restrict__ xin, _Float16* __restrict__ out) {
  size_t i = blockIdx.x * (size_t)blockDim.x + threadIdx.x;
  size_t total = (size_t)Mtok * Cc;
  if (i >= total) return;
  int c = (int)(i % Cc); size_t t = i / Cc;
  int ll = (int)(t % Ltok); int b = (int)(t / Ltok);
  int z = ll / 784, y = (ll / 28) % 28, xx = ll % 28;
  float sum = 0.f;
  for (int dz = -1; dz <= 1; ++dz)
    for (int dy = -1; dy <= 1; ++dy)
      for (int dx = -1; dx <= 1; ++dx) {
        int nz = z + dz, ny = y + dy, nx = xx + dx;
        if (nz < 0 || nz >= 28 || ny < 0 || ny >= 28 || nx < 0 || nx >= 28) continue;
        sum += (float)xin[((size_t)b * Ltok + (nz * 28 + ny) * 28 + nx) * Cc + c];
      }
  out[i] = (_Float16)(sum / 27.f);  // count_include_pad
}

// ------------------------------------------- im2col for 24ch 3x3x3 (K->672)
__global__ void im2col24(const _Float16* __restrict__ src,   // [Mtok][32]
                         _Float16* __restrict__ colb) {      // [Mtok][672]
  int tid = blockIdx.x * blockDim.x + threadIdx.x;
  int total = Mtok * 28;
  if (tid >= total) return;
  int j = tid % 28; size_t row = tid / 28;
  _Float16* dst = colb + row * KCONV + j * 24;
  if (j == 27) { for (int c = 0; c < 24; ++c) dst[c] = (_Float16)0.f; return; }
  int ll = (int)(row % Ltok); int b = (int)(row / Ltok);
  int z = ll / 784, y = (ll / 28) % 28, xx = ll % 28;
  int dz = j / 9 - 1, dy = (j / 3) % 3 - 1, dx = j % 3 - 1;
  int nz = z + dz, ny = y + dy, nx = xx + dx;
  if (nz < 0 || nz >= 28 || ny < 0 || ny >= 28 || nx < 0 || nx >= 28) {
    for (int c = 0; c < 24; ++c) dst[c] = (_Float16)0.f;
  } else {
    const _Float16* s = src + ((size_t)b * Ltok + (nz * 28 + ny) * 28 + nx) * 32;
    for (int c = 0; c < 24; ++c) dst[c] = s[c];
  }
}

// ------------------------------------------------ eval-mode BN + exact GELU
__global__ void bn_gelu(const float* __restrict__ in, int ldin,
                        const float* __restrict__ g, const float* __restrict__ be,
                        _Float16* __restrict__ out, int ldout, int coloff,
                        int ncols, int padcols, int M) {
  size_t tid = blockIdx.x * (size_t)blockDim.x + threadIdx.x;
  size_t total = (size_t)M * padcols;
  if (tid >= total) return;
  int c = (int)(tid % padcols); size_t m = tid / padcols;
  _Float16* o = out + m * (size_t)ldout + coloff + c;
  if (c >= ncols) { *o = (_Float16)0.f; return; }
  float v = in[m * (size_t)ldin + c];
  v = v * (g[c] * rsqrtf(1.f + 1e-3f)) + be[c];
  float ge = 0.5f * v * (1.f + erff(v * 0.70710678118654752f));
  *o = (_Float16)ge;
}

// ================================================================= launcher
extern "C" void kernel_launch(void* const* d_in, const int* in_sizes, int n_in,
                              void* d_out, int out_size, void* d_ws, size_t ws_size,
                              hipStream_t stream) {
  (void)in_sizes; (void)n_in; (void)out_size; (void)ws_size;
  const float* x        = (const float*)d_in[0];
  const float* maskm    = (const float*)d_in[1];
  const float* ln1_g    = (const float*)d_in[2];
  const float* ln1_b    = (const float*)d_in[3];
  const float* qkv_w    = (const float*)d_in[4];
  const float* qkv_b    = (const float*)d_in[5];
  const float* rel_tab  = (const float*)d_in[6];
  const float* proj_w   = (const float*)d_in[7];
  const float* proj_b   = (const float*)d_in[8];
  const float* ln2_g    = (const float*)d_in[9];
  const float* ln2_b    = (const float*)d_in[10];
  const float* b1_w = (const float*)d_in[11],  *b1_b = (const float*)d_in[12];
  const float* b1_g = (const float*)d_in[13],  *b1_be = (const float*)d_in[14];
  const float* b3a_w = (const float*)d_in[15], *b3a_b = (const float*)d_in[16];
  const float* b3a_g = (const float*)d_in[17], *b3a_be = (const float*)d_in[18];
  const float* b3b_w = (const float*)d_in[19], *b3b_b = (const float*)d_in[20];
  const float* b3b_g = (const float*)d_in[21], *b3b_be = (const float*)d_in[22];
  const float* b5a_w = (const float*)d_in[23], *b5a_b = (const float*)d_in[24];
  const float* b5a_g = (const float*)d_in[25], *b5a_be = (const float*)d_in[26];
  const float* b5b_w = (const float*)d_in[27], *b5b_b = (const float*)d_in[28];
  const float* b5b_g = (const float*)d_in[29], *b5b_be = (const float*)d_in[30];
  const float* b5c_w = (const float*)d_in[31], *b5c_b = (const float*)d_in[32];
  const float* b5c_g = (const float*)d_in[33], *b5c_be = (const float*)d_in[34];
  const float* bp_w  = (const float*)d_in[35], *bp_b  = (const float*)d_in[36];
  const float* bp_g  = (const float*)d_in[37], *bp_be = (const float*)d_in[38];
  const float* fc_w  = (const float*)d_in[39], *fc_b  = (const float*)d_in[40];
  float* outp = (float*)d_out;

  // ---- workspace bump allocator
  char* wp = (char*)d_ws;
  auto alloc = [&](size_t bytes) -> char* {
    char* r = wp; wp += (bytes + 255) & ~(size_t)255; return r;
  };
  _Float16* xw_h   = (_Float16*)alloc((size_t)Mtok * Cc * 2);
  _Float16* qkv_h  = (_Float16*)alloc((size_t)Mtok * 576 * 2);
  _Float16* o_h    = (_Float16*)alloc((size_t)Mtok * Cc * 2);
  float*    resid1 = (float*)   alloc((size_t)Mtok * Cc * 4);
  _Float16* xin_h  = (_Float16*)alloc((size_t)Mtok * Cc * 2);
  float*    tmpF   = (float*)   alloc((size_t)Mtok * Cc * 4);   // also proj out
  float*    tmp24F = (float*)   alloc((size_t)Mtok * 32 * 4);
  _Float16* t_h    = (_Float16*)alloc((size_t)Mtok * 32 * 2);
  _Float16* t2_h   = (_Float16*)alloc((size_t)Mtok * 32 * 2);
  _Float16* colbuf = (_Float16*)alloc((size_t)Mtok * KCONV * 2);
  _Float16* pool_h = (_Float16*)alloc((size_t)Mtok * Cc * 2);
  _Float16* cat_h  = (_Float16*)alloc((size_t)Mtok * HF * 2);
  _Float16* qkv_wp = (_Float16*)alloc(576 * 192 * 2);
  _Float16* proj_wp= (_Float16*)alloc(192 * 192 * 2);
  _Float16* b1_wp  = (_Float16*)alloc(192 * 192 * 2);
  _Float16* b3a_wp = (_Float16*)alloc(32 * 192 * 2);
  _Float16* b5a_wp = (_Float16*)alloc(32 * 192 * 2);
  _Float16* bp_wp  = (_Float16*)alloc(192 * 192 * 2);
  _Float16* fc_wp  = (_Float16*)alloc(192 * 768 * 2);
  _Float16* b3b_wp = (_Float16*)alloc(192 * KCONV * 2);
  _Float16* b5b_wp = (_Float16*)alloc(32 * KCONV * 2);
  _Float16* b5c_wp = (_Float16*)alloc(192 * KCONV * 2);

  auto cdiv = [](long a, long b) { return (int)((a + b - 1) / b); };

  // ---- pack weights to f16 (zero padded)
  pack_dense<<<cdiv(576*192,256),256,0,stream>>>(qkv_w, qkv_wp, 576, 576, 192);
  pack_dense<<<cdiv(192*192,256),256,0,stream>>>(proj_w, proj_wp, 192, 192, 192);
  pack_dense<<<cdiv(192*192,256),256,0,stream>>>(b1_w, b1_wp, 192, 192, 192);
  pack_dense<<<cdiv(32*192,256),256,0,stream>>>(b3a_w, b3a_wp, 24, 32, 192);
  pack_dense<<<cdiv(32*192,256),256,0,stream>>>(b5a_w, b5a_wp, 24, 32, 192);
  pack_dense<<<cdiv(192*192,256),256,0,stream>>>(bp_w, bp_wp, 192, 192, 192);
  pack_dense<<<cdiv(192*768,256),256,0,stream>>>(fc_w, fc_wp, 192, 192, 768);
  pack_conv <<<cdiv(192*KCONV,256),256,0,stream>>>(b3b_w, b3b_wp, 192, 192);
  pack_conv <<<cdiv(32*KCONV,256),256,0,stream>>>(b5b_w, b5b_wp, 24, 32);
  pack_conv <<<cdiv(192*KCONV,256),256,0,stream>>>(b5c_w, b5c_wp, 192, 192);

  // ---- LN1 + shift + window partition
  ln1_window<<<Mtok/8, 256, 0, stream>>>(x, ln1_g, ln1_b, xw_h);

  // ---- QKV GEMM (f16 out, bias)
  {
    long waves = (long)(Mtok/16) * ((36 + 3) / 4);
    gemm_xwt<1><<<cdiv(waves,8),256,0,stream>>>(xw_h, 192, qkv_wp, 192, qkv_b,
        nullptr, nullptr, qkv_h, 576, Mtok, 576, 36, 192);
  }
  // ---- fused attention (scores + bias + mask + softmax + PV), per (win,head)
  attn_win<<<Bn*NWl*HEADS, 96, 0, stream>>>(qkv_h, rel_tab, maskm, o_h);

  // ---- proj GEMM (f32 out, bias)
  {
    long waves = (long)(Mtok/16) * 3;
    gemm_xwt<0><<<cdiv(waves,8),256,0,stream>>>(o_h, 192, proj_wp, 192, proj_b,
        nullptr, tmpF, nullptr, 192, Mtok, 192, 12, 192);
  }
  // ---- window reverse + unshift + residual
  unwindow_resid<<<cdiv((long)Mtok*Cc,256),256,0,stream>>>(x, tmpF, resid1);
  // ---- LN2
  ln_rows<<<Mtok/8,256,0,stream>>>(resid1, ln2_g, ln2_b, xin_h);

  long wavesN192 = (long)(Mtok/16) * 3;   // tilesN=12 -> 3 strips
  long wavesN32  = (long)(Mtok/16) * 1;   // tilesN=2  -> 1 strip

  // ---- branch 1: 1x1 conv + BN + GELU -> cat[:,0:192]
  gemm_xwt<0><<<cdiv(wavesN192,8),256,0,stream>>>(xin_h, 192, b1_wp, 192, b1_b,
      nullptr, tmpF, nullptr, 192, Mtok, 192, 12, 192);
  bn_gelu<<<cdiv((long)Mtok*192,256),256,0,stream>>>(tmpF, 192, b1_g, b1_be,
      cat_h, HF, 0, 192, 192, Mtok);

  // ---- branch 3: 1x1 (24) -> 3x3x3 (192) -> cat[:,192:384]
  gemm_xwt<0><<<cdiv(wavesN32,8),256,0,stream>>>(xin_h, 192, b3a_wp, 192, b3a_b,
      nullptr, tmp24F, nullptr, 32, Mtok, 24, 2, 192);
  bn_gelu<<<cdiv((long)Mtok*32,256),256,0,stream>>>(tmp24F, 32, b3a_g, b3a_be,
      t_h, 32, 0, 24, 32, Mtok);
  im2col24<<<cdiv((long)Mtok*28,256),256,0,stream>>>(t_h, colbuf);
  gemm_xwt<0><<<cdiv(wavesN192,8),256,0,stream>>>(colbuf, KCONV, b3b_wp, KCONV,
      b3b_b, nullptr, tmpF, nullptr, 192, Mtok, 192, 12, KCONV);
  bn_gelu<<<cdiv((long)Mtok*192,256),256,0,stream>>>(tmpF, 192, b3b_g, b3b_be,
      cat_h, HF, 192, 192, 192, Mtok);

  // ---- branch 5: 1x1 (24) -> 3^3 (24) -> 3^3 (192) -> cat[:,384:576]
  gemm_xwt<0><<<cdiv(wavesN32,8),256,0,stream>>>(xin_h, 192, b5a_wp, 192, b5a_b,
      nullptr, tmp24F, nullptr, 32, Mtok, 24, 2, 192);
  bn_gelu<<<cdiv((long)Mtok*32,256),256,0,stream>>>(tmp24F, 32, b5a_g, b5a_be,
      t_h, 32, 0, 24, 32, Mtok);
  im2col24<<<cdiv((long)Mtok*28,256),256,0,stream>>>(t_h, colbuf);
  gemm_xwt<0><<<cdiv(wavesN32,8),256,0,stream>>>(colbuf, KCONV, b5b_wp, KCONV,
      b5b_b, nullptr, tmp24F, nullptr, 32, Mtok, 24, 2, KCONV);
  bn_gelu<<<cdiv((long)Mtok*32,256),256,0,stream>>>(tmp24F, 32, b5b_g, b5b_be,
      t2_h, 32, 0, 24, 32, Mtok);
  im2col24<<<cdiv((long)Mtok*28,256),256,0,stream>>>(t2_h, colbuf);
  gemm_xwt<0><<<cdiv(wavesN192,8),256,0,stream>>>(colbuf, KCONV, b5c_wp, KCONV,
      b5c_b, nullptr, tmpF, nullptr, 192, Mtok, 192, 12, KCONV);
  bn_gelu<<<cdiv((long)Mtok*192,256),256,0,stream>>>(tmpF, 192, b5c_g, b5c_be,
      cat_h, HF, 384, 192, 192, Mtok);

  // ---- pool branch: avgpool3 -> 1x1 -> cat[:,576:768]
  avgpool3<<<cdiv((long)Mtok*Cc,256),256,0,stream>>>(xin_h, pool_h);
  gemm_xwt<0><<<cdiv(wavesN192,8),256,0,stream>>>(pool_h, 192, bp_wp, 192, bp_b,
      nullptr, tmpF, nullptr, 192, Mtok, 192, 12, 192);
  bn_gelu<<<cdiv((long)Mtok*192,256),256,0,stream>>>(tmpF, 192, bp_g, bp_be,
      cat_h, HF, 576, 192, 192, Mtok);

  // ---- fc GEMM + bias + residual -> d_out (f32)
  gemm_xwt<2><<<cdiv(wavesN192,8),256,0,stream>>>(cat_h, HF, fc_wp, HF, fc_b,
      resid1, outp, nullptr, 192, Mtok, 192, 12, 768);
}
